// EdgeGroupEmbeddingGenerator_34256659153225
// MI455X (gfx1250) — matrix operand order repaired and checked by reading, not compiled, not run on hardware
//
#include <hip/hip_runtime.h>
#include <hip/hip_bf16.h>
#include <math.h>

typedef __attribute__((ext_vector_type(16))) _Float16 v16h;
typedef __attribute__((ext_vector_type(8)))  float    v8f;

#define EDGE_DIM 128
#define HIDDEN   128
#define LN_EPS   1e-5f

// Branch-free erf (Abramowitz & Stegun 7.1.26, |err| <= 1.5e-7).
// Avoids libm erff's EXEC-divergent path; uses hw v_exp_f32 + fast rcp,
// which co-execute with the WMMA pipe.
__device__ __forceinline__ float erf_fast(float x) {
    float ax = fabsf(x);
    float t  = __fdividef(1.0f, fmaf(0.3275911f, ax, 1.0f));
    float p  = fmaf(t, 1.061405429f, -1.453152027f);
    p = fmaf(t, p, 1.421413741f);
    p = fmaf(t, p, -0.284496736f);
    p = fmaf(t, p, 0.254829592f);
    float r = 1.0f - t * p * __expf(-ax * ax);
    return copysignf(r, x);
}

__device__ __forceinline__ float gelu_exact(float x) {
    return 0.5f * x * (1.0f + erf_fast(x * 0.70710678118654752f));
}

// Index into the swizzled f16 A-fragment LDS buffer for WMMA 16x16x32 f16.
// Per ISA 7.12.2 (16-bit A 16x32): lanes 0-15 hold K {0..7,16..23}, lanes 16-31
// hold K {8..15,24..31}; 16 halves per lane per 32-K chunk.
__device__ __forceinline__ int a_swz_index(int m, int k) {
    int kc   = k >> 5;
    int kl   = k & 31;
    int lane = m + (((kl >> 3) & 1) << 4);
    int e    = (kl & 7) + ((kl >> 4) << 3);
    return (kc * 32 + lane) * 16 + e;
}

// Pack fp32 row-major (K x N) weights into f16 WMMA B-fragment order:
// fragment (kc, nt) is 32 lanes x 16 halves; lane L covers N = nt*16 + (L&15),
// K = kc*32 + (L<16 ? 0 : 8) + (e&7) + (e<8 ? 0 : 16) for e in [0,16).
__global__ void pack_weights(const float* __restrict__ src,
                             _Float16* __restrict__ dst,
                             int N, int total) {
    int idx = blockIdx.x * blockDim.x + threadIdx.x;
    if (idx >= total) return;
    int e   = idx & 15;
    int L   = (idx >> 4) & 31;
    int ntl = N >> 4;
    int nt  = (idx >> 9) % ntl;
    int kc  = idx / (512 * ntl);
    int k   = kc * 32 + ((L & 16) ? 8 : 0) + (e & 7) + ((e & 8) ? 16 : 0);
    int n   = nt * 16 + (L & 15);
    dst[idx] = (_Float16)src[k * N + n];
}

__global__ void __launch_bounds__(256)
edge_group_mlp(const float* __restrict__ emb,
               const int*   __restrict__ groups,
               const float* __restrict__ scores,
               const _Float16* __restrict__ W1p,
               const float* __restrict__ b1,
               const float* __restrict__ g1,
               const float* __restrict__ be1,
               const _Float16* __restrict__ W2p,
               const float* __restrict__ b2,
               const float* __restrict__ g2,
               const float* __restrict__ be2,
               const _Float16* __restrict__ Wmp,
               const float* __restrict__ Wm,   // fp32 (129x128): last row used directly
               const float* __restrict__ bm,
               float* __restrict__ out,
               int G) {
    __shared__ __align__(32) _Float16 ldsA[8 * 32 * 16]; // swizzled A frags (8 KB)
    __shared__ float ldsH[16 * 256];                     // activation staging (16 KB)
    __shared__ float redS[256];
    __shared__ float redQ[256];
    __shared__ float stat[32];

    const int tid  = threadIdx.x;
    const int lane = tid & 31;
    const int wave = tid >> 5;
    const int g0   = blockIdx.x * 16;

    // ---- Stage concat(emb_i, emb_j) (16 x 256) as f16 in A-fragment swizzle
    {
        int m   = tid >> 4;
        int seg = tid & 15;
        int g   = g0 + m;
        int gg  = (g < G) ? g : (G - 1);
        int i0  = groups[2 * gg + 0];
        int i1  = groups[2 * gg + 1];
        const float* s0 = emb + (size_t)i0 * EDGE_DIM;
        const float* s1 = emb + (size_t)i1 * EDGE_DIM;
        int kbase = seg * 16;
#pragma unroll
        for (int j = 0; j < 16; ++j) {
            int k = kbase + j;
            float v = (k < EDGE_DIM) ? s0[k] : s1[k - EDGE_DIM];
            ldsA[a_swz_index(m, k)] = (_Float16)v;
        }
    }
    __syncthreads();

    // ---- Layer 1: (16x256) @ (256x256), bias + exact GELU -> ldsH (16x256)
#pragma unroll
    for (int t2 = 0; t2 < 2; ++t2) {
        int nt = wave * 2 + t2;
        v8f acc = {};
#pragma unroll
        for (int kc = 0; kc < 8; ++kc) {
            v16h a = *(const v16h*)(&ldsA[(kc * 32 + lane) * 16]);
            v16h b = *(const v16h*)(W1p + (((kc * 16 + nt) * 32 + lane) << 4));
            if (kc < 7)
                __builtin_prefetch(W1p + ((((kc + 1) * 16 + nt) * 32 + lane) << 4), 0, 1);
            acc = __builtin_amdgcn_wmma_f32_16x16x32_f16(false, a, false, b,
                                                         (short)0, acc, false, false);
        }
        int n     = nt * 16 + (lane & 15);
        int mbase = (lane >> 4) << 3;
        float bb  = b1[n];
#pragma unroll
        for (int r = 0; r < 8; ++r)
            ldsH[(mbase + r) * 256 + n] = gelu_exact(acc[r] + bb);
    }
    __syncthreads();

    // ---- LayerNorm over 256, write f16 A-fragments back to ldsA
    {
        int row = tid >> 4, seg = tid & 15;
        float s = 0.f, q = 0.f;
        int base = row * 256 + seg * 16;
#pragma unroll
        for (int j = 0; j < 16; ++j) { float v = ldsH[base + j]; s += v; q += v * v; }
        redS[tid] = s; redQ[tid] = q;
    }
    __syncthreads();
    if (tid < 16) {
        float s = 0.f, q = 0.f;
#pragma unroll
        for (int j = 0; j < 16; ++j) { s += redS[tid * 16 + j]; q += redQ[tid * 16 + j]; }
        float mu  = s * (1.0f / 256.0f);
        float var = q * (1.0f / 256.0f) - mu * mu;
        stat[tid * 2]     = mu;
        stat[tid * 2 + 1] = rsqrtf(var + LN_EPS);
    }
    __syncthreads();
    {
        int row = tid >> 4, seg = tid & 15;
        float mu = stat[row * 2], rs = stat[row * 2 + 1];
#pragma unroll
        for (int j = 0; j < 16; ++j) {
            int n = seg * 16 + j;
            float v = (ldsH[row * 256 + n] - mu) * rs * g1[n] + be1[n];
            ldsA[a_swz_index(row, n)] = (_Float16)v;
        }
    }
    __syncthreads();

    // ---- Layer 2: (16x256) @ (256x128), bias + ReLU -> ldsH (16x128)
    {
        int nt = wave;
        v8f acc = {};
#pragma unroll
        for (int kc = 0; kc < 8; ++kc) {
            v16h a = *(const v16h*)(&ldsA[(kc * 32 + lane) * 16]);
            v16h b = *(const v16h*)(W2p + (((kc * 8 + nt) * 32 + lane) << 4));
            acc = __builtin_amdgcn_wmma_f32_16x16x32_f16(false, a, false, b,
                                                         (short)0, acc, false, false);
        }
        int n     = nt * 16 + (lane & 15);
        int mbase = (lane >> 4) << 3;
        float bb  = b2[n];
#pragma unroll
        for (int r = 0; r < 8; ++r) {
            float v = acc[r] + bb;
            ldsH[(mbase + r) * 128 + n] = v > 0.f ? v : 0.f;
        }
    }
    __syncthreads();

    // ---- LayerNorm over 128, write f16 A-fragments (4 K-chunks) to ldsA
    {
        int row = tid >> 4, seg = tid & 15;
        float s = 0.f, q = 0.f;
        int base = row * 128 + seg * 8;
#pragma unroll
        for (int j = 0; j < 8; ++j) { float v = ldsH[base + j]; s += v; q += v * v; }
        redS[tid] = s; redQ[tid] = q;
    }
    __syncthreads();
    if (tid < 16) {
        float s = 0.f, q = 0.f;
#pragma unroll
        for (int j = 0; j < 16; ++j) { s += redS[tid * 16 + j]; q += redQ[tid * 16 + j]; }
        float mu  = s * (1.0f / 128.0f);
        float var = q * (1.0f / 128.0f) - mu * mu;
        stat[tid * 2]     = mu;
        stat[tid * 2 + 1] = rsqrtf(var + LN_EPS);
    }
    __syncthreads();
    {
        int row = tid >> 4, seg = tid & 15;
        float mu = stat[row * 2], rs = stat[row * 2 + 1];
#pragma unroll
        for (int j = 0; j < 8; ++j) {
            int n = seg * 8 + j;
            float v = (ldsH[row * 128 + n] - mu) * rs * g2[n] + be2[n];
            ldsA[a_swz_index(row, n)] = (_Float16)v;
        }
    }
    __syncthreads();

    // ---- Layer 3: (16x128) @ (128x128) + score * Wm[128,:] + bm, ReLU, store
    {
        int nt = wave;
        v8f acc = {};
#pragma unroll
        for (int kc = 0; kc < 4; ++kc) {
            v16h a = *(const v16h*)(&ldsA[(kc * 32 + lane) * 16]);
            v16h b = *(const v16h*)(Wmp + (((kc * 8 + nt) * 32 + lane) << 4));
            acc = __builtin_amdgcn_wmma_f32_16x16x32_f16(false, a, false, b,
                                                         (short)0, acc, false, false);
        }
        int n     = nt * 16 + (lane & 15);
        int mbase = (lane >> 4) << 3;
        float bb  = bm[n];
        float wl  = Wm[128 * 128 + n];   // importance column of Wm
#pragma unroll
        for (int r = 0; r < 8; ++r) {
            int g = g0 + mbase + r;
            if (g < G) {
                float v = acc[r] + bb + scores[g] * wl;
                out[(size_t)g * HIDDEN + n] = v > 0.f ? v : 0.f;
            }
        }
    }
}

extern "C" void kernel_launch(void* const* d_in, const int* in_sizes, int n_in,
                              void* d_out, int out_size, void* d_ws, size_t ws_size,
                              hipStream_t stream) {
    const float* emb    = (const float*)d_in[0];
    const int*   groups = (const int*)d_in[1];
    const float* scores = (const float*)d_in[2];
    const float* W1     = (const float*)d_in[3];
    const float* b1     = (const float*)d_in[4];
    const float* g1     = (const float*)d_in[5];
    const float* be1    = (const float*)d_in[6];
    const float* W2     = (const float*)d_in[7];
    const float* b2     = (const float*)d_in[8];
    const float* g2     = (const float*)d_in[9];
    const float* be2    = (const float*)d_in[10];
    const float* Wm     = (const float*)d_in[11];
    const float* bm     = (const float*)d_in[12];
    float* out = (float*)d_out;
    int G = in_sizes[2];   // importance_scores count == number of groups

    char* ws = (char*)d_ws;
    _Float16* W1p = (_Float16*)ws;                       // 65536 halves
    _Float16* W2p = (_Float16*)(ws + 131072);            // 32768 halves
    _Float16* Wmp = (_Float16*)(ws + 131072 + 65536);    // 16384 halves

    pack_weights<<<256, 256, 0, stream>>>(W1, W1p, 256, 65536);
    pack_weights<<<128, 256, 0, stream>>>(W2, W2p, 128, 32768);
    pack_weights<<< 64, 256, 0, stream>>>(Wm, Wmp, 128, 16384);  // first 128 rows of Wm

    int blocks = (G + 15) / 16;
    edge_group_mlp<<<blocks, 256, 0, stream>>>(emb, groups, scores,
                                               W1p, b1, g1, be1,
                                               W2p, b2, g2, be2,
                                               Wmp, Wm, bm, out, G);
}
